// GNN_14465449853061
// MI455X (gfx1250) — matrix operand order repaired and checked by reading, not compiled, not run on hardware
//
#include <hip/hip_runtime.h>
#include <hip/hip_bf16.h>

// ---------- WMMA vector types (CDNA5 gfx1250, wave32) ----------
typedef __attribute__((ext_vector_type(16))) __bf16 v16bf;
typedef __attribute__((ext_vector_type(8)))  float  v8f;

#define TPB 256

// ---------------------------------------------------------------
// Zero a float region (grid-stride).
__global__ void k_zero(float* __restrict__ p, long long n) {
    long long i = (long long)blockIdx.x * blockDim.x + threadIdx.x;
    long long stride = (long long)gridDim.x * blockDim.x;
    for (; i < n; i += stride) p[i] = 0.0f;
}

// ---------------------------------------------------------------
// Convert 4 weight matrices W[k][n] (f32, 128x128) -> bf16 transposed WT[n][k],
// packed contiguously: [W1lT][W1rT][W2lT][W2rT].
__global__ void k_convert_weights(const float* __restrict__ W1l,
                                  const float* __restrict__ W1r,
                                  const float* __restrict__ W2l,
                                  const float* __restrict__ W2r,
                                  __bf16* __restrict__ outAll) {
    int idx = blockIdx.x * TPB + threadIdx.x;   // 0 .. 65535
    int m = idx >> 14;
    int r = idx & 16383;
    int n = r >> 7;
    int k = r & 127;
    const float* W = (m == 0) ? W1l : (m == 1) ? W1r : (m == 2) ? W2l : W2r;
    outAll[m * 16384 + n * 128 + k] = (__bf16)W[k * 128 + n];
}

// ---------------------------------------------------------------
// Degree: one thread per edge.
__global__ void k_degree(const int* __restrict__ dst, float* __restrict__ deg, int E) {
    int e = blockIdx.x * TPB + threadIdx.x;
    if (e >= E) return;
    unsafeAtomicAdd(&deg[dst[e]], 1.0f);
}

// ---------------------------------------------------------------
// Scatter-add: one wave (32 lanes) per edge, each lane handles float4.
// 512B coalesced gather per edge, hardware f32 atomics for scatter.
__global__ void k_scatter(const float* __restrict__ feat,
                          const int* __restrict__ src,
                          const int* __restrict__ dst,
                          float* agg, int E) {
    long long tid = (long long)blockIdx.x * TPB + threadIdx.x;
    int e = (int)(tid >> 5);
    if (e >= E) return;
    int g = ((int)tid & 31) * 4;
    int s = src[e];
    int d = dst[e];
    if (e + 2048 < E) __builtin_prefetch(src + e + 2048, 0, 1);
    float4 v = *(const float4*)(feat + (size_t)s * 128 + g);
    float* o = agg + (size_t)d * 128 + g;
    unsafeAtomicAdd(o + 0, v.x);
    unsafeAtomicAdd(o + 1, v.y);
    unsafeAtomicAdd(o + 2, v.z);
    unsafeAtomicAdd(o + 3, v.w);
}

// ---------------------------------------------------------------
// Pack 16 scaled f32 values into a v16bf A-fragment segment-wise.
// Contiguous k order used here: elements 0..15 = K seg0..seg0+15 (B matches).
__device__ inline v16bf pack_a(float4 q0, float4 q1, float4 q2, float4 q3, float s) {
    v16bf a;
    a[0]  = (__bf16)(q0.x * s); a[1]  = (__bf16)(q0.y * s);
    a[2]  = (__bf16)(q0.z * s); a[3]  = (__bf16)(q0.w * s);
    a[4]  = (__bf16)(q1.x * s); a[5]  = (__bf16)(q1.y * s);
    a[6]  = (__bf16)(q1.z * s); a[7]  = (__bf16)(q1.w * s);
    a[8]  = (__bf16)(q2.x * s); a[9]  = (__bf16)(q2.y * s);
    a[10] = (__bf16)(q2.z * s); a[11] = (__bf16)(q2.w * s);
    a[12] = (__bf16)(q3.x * s); a[13] = (__bf16)(q3.y * s);
    a[14] = (__bf16)(q3.z * s); a[15] = (__bf16)(q3.w * s);
    return a;
}

// ---------------------------------------------------------------
// Fused SAGE GEMM: out = relu( (agg/deg) @ Wl + X @ Wr + b ), all dims 128.
// wBase: [WlT (16384 bf16)][WrT (16384 bf16)], WT[n][k] layout.
// 8 waves/block, 16-row strip per wave. A fragments fully resident in regs
// (64 VGPRs for both matrices), col-tile-outer loop with one 8-VGPR acc.
// Relu is baked in; row-bounds check is wave-uniform (fast path unguarded).
__global__ void __launch_bounds__(TPB)
k_sage_gemm(const float* A1 /*agg*/, const float* deg, const float* A2 /*x|h*/,
            const __bf16* __restrict__ wBase, const float* __restrict__ bias,
            float* out, int nrows) {
    __shared__ __align__(32) __bf16 smem[2 * 128 * 128];   // 64 KB: WlT | WrT

    // cooperative weight load: 65536 B = 4096 uint4
    {
        const uint4* gsrc = (const uint4*)wBase;
        uint4* gdst = (uint4*)smem;
        for (int i = threadIdx.x; i < 4096; i += TPB) gdst[i] = gsrc[i];
    }
    __syncthreads();

    const int wave = threadIdx.x >> 5;
    const int lane = threadIdx.x & 31;
    const int half = lane >> 4;           // 0: lanes 0-15, 1: lanes 16-31
    const int l15  = lane & 15;

    const int strip = blockIdx.x * 8 + wave;
    const int strips_total = (nrows + 15) >> 4;
    if (strip >= strips_total) return;

    const bool full_strip = (strip * 16 + 16 <= nrows);   // wave-uniform

    int row = strip * 16 + l15;
    int rowc = row < nrows ? row : nrows - 1;
    float scale = 1.0f / fmaxf(deg[rowc], 1.0f);

    const float4* pa1 = (const float4*)(A1 + (size_t)rowc * 128);
    const float4* pa2 = (const float4*)(A2 + (size_t)rowc * 128);
    const __bf16* sWl = smem;
    const __bf16* sWr = smem + 128 * 128;

    // ---- load ALL A fragments up front, convert to bf16 immediately ----
    // A layout per k-chunk: lane<16 covers K = k0..k0+15 ;
    // lane>=16 covers k0+16..k0+31.
    v16bf a1[4], a2[4];
#pragma unroll
    for (int kc = 0; kc < 4; ++kc) {
        int i0 = kc * 8 + half * 4;       // float4 index of seg0
        float4 q0 = pa1[i0], q1 = pa1[i0 + 1], q2 = pa1[i0 + 2], q3 = pa1[i0 + 3];
        float4 r0 = pa2[i0], r1 = pa2[i0 + 1], r2 = pa2[i0 + 2], r3 = pa2[i0 + 3];
        a1[kc] = pack_a(q0, q1, q2, q3, scale);
        a2[kc] = pack_a(r0, r1, r2, r3, 1.0f);
    }

    // ---- col-tile-outer WMMA loop, one accumulator tile at a time ----
#pragma unroll
    for (int ct = 0; ct < 8; ++ct) {
        int col = ct * 16 + l15;
        // B layout: lanes 0-15 -> K 0..15 of column=col, lanes 16-31 -> K 16..31
        const __bf16* pl = sWl + col * 128 + half * 16;
        const __bf16* pr = sWr + col * 128 + half * 16;

        v8f acc = (v8f)(0.0f);
#pragma unroll
        for (int kc = 0; kc < 4; ++kc) {
            v16bf bl = *(const v16bf*)(pl + kc * 32);
            acc = __builtin_amdgcn_wmma_f32_16x16x32_bf16(
                false, a1[kc], false, bl, (short)0, acc, false, false);
        }
#pragma unroll
        for (int kc = 0; kc < 4; ++kc) {
            v16bf br = *(const v16bf*)(pr + kc * 32);
            acc = __builtin_amdgcn_wmma_f32_16x16x32_bf16(
                false, a2[kc], false, br, (short)0, acc, false, false);
        }

        // C/D layout: VGPR r -> (M = strip*16 + half*8 + r, N = col)
        float bv = bias[col];
        float* po = out + (size_t)(strip * 16 + half * 8) * 128 + col;
        if (full_strip) {
#pragma unroll
            for (int r = 0; r < 8; ++r)
                po[(size_t)r * 128] = fmaxf(acc[r] + bv, 0.0f);
        } else {
#pragma unroll
            for (int r = 0; r < 8; ++r) {
                int m = strip * 16 + half * 8 + r;
                if (m < nrows) po[(size_t)r * 128] = fmaxf(acc[r] + bv, 0.0f);
            }
        }
    }
}

// ---------------------------------------------------------------
// Final head: out[n][2] = h[n] @ Wlin + blin. Wlin is [128][2].
__global__ void k_final(const float* __restrict__ h, const float* __restrict__ Wlin,
                        const float* __restrict__ blin, float* __restrict__ out, int n) {
    __shared__ float w[256];
    w[threadIdx.x] = Wlin[threadIdx.x & 255];
    __syncthreads();
    int i = blockIdx.x * TPB + threadIdx.x;
    if (i >= n) return;
    const float4* p = (const float4*)(h + (size_t)i * 128);
    float s0 = blin[0], s1 = blin[1];
#pragma unroll 8
    for (int k4 = 0; k4 < 32; ++k4) {
        float4 v = p[k4];
        int k = k4 * 4;
        s0 += v.x * w[(k + 0) * 2] + v.y * w[(k + 1) * 2] + v.z * w[(k + 2) * 2] + v.w * w[(k + 3) * 2];
        s1 += v.x * w[(k + 0) * 2 + 1] + v.y * w[(k + 1) * 2 + 1] + v.z * w[(k + 2) * 2 + 1] + v.w * w[(k + 3) * 2 + 1];
    }
    out[(size_t)i * 2 + 0] = s0;
    out[(size_t)i * 2 + 1] = s1;
}

// ---------------------------------------------------------------
extern "C" void kernel_launch(void* const* d_in, const int* in_sizes, int n_in,
                              void* d_out, int out_size, void* d_ws, size_t ws_size,
                              hipStream_t stream) {
    const float* x    = (const float*)d_in[0];
    const int*   ei   = (const int*)d_in[1];
    const float* W1l  = (const float*)d_in[2];
    const float* b1   = (const float*)d_in[3];
    const float* W1r  = (const float*)d_in[4];
    const float* W2l  = (const float*)d_in[5];
    const float* b2   = (const float*)d_in[6];
    const float* W2r  = (const float*)d_in[7];
    const float* Wlin = (const float*)d_in[8];
    const float* blin = (const float*)d_in[9];

    const int N = in_sizes[0] / 128;
    const int E = in_sizes[1] / 2;
    const int* esrc = ei;
    const int* edst = ei + E;

    // workspace layout
    char* ws = (char*)d_ws;
    auto align256 = [](size_t v) { return (v + 255) & ~(size_t)255; };
    __bf16* wT = (__bf16*)ws;                         // 4*16384 bf16 = 131072 B
    size_t off = 131072;
    float* deg = (float*)(ws + off);
    off = align256(off + (size_t)N * 4);
    float* agg = (float*)(ws + off);
    size_t off_agg = off;
    off = align256(off + (size_t)N * 512);
    float* hbuf = (float*)(ws + off);                 // h1, overwritten by h2 in place

    const long long aggN = (long long)N * 128;
    const int strips = (N + 15) / 16;
    const int gemm_blocks = (strips + 7) / 8;
    const long long scatter_threads = (long long)E * 32;
    const int scatter_blocks = (int)((scatter_threads + TPB - 1) / TPB);

    // 1) weights -> bf16 transposed
    k_convert_weights<<<256, TPB, 0, stream>>>(W1l, W1r, W2l, W2r, wT);

    // 2) zero deg + agg (contiguous region)
    {
        long long zn = (long long)((off_agg + (size_t)N * 512 - 131072) / 4);
        k_zero<<<4096, TPB, 0, stream>>>(deg, zn);
    }

    // 3) degrees
    k_degree<<<(E + TPB - 1) / TPB, TPB, 0, stream>>>(edst, deg, E);

    // 4) layer-1 scatter: x -> agg
    k_scatter<<<scatter_blocks, TPB, 0, stream>>>(x, esrc, edst, agg, E);

    // 5) layer-1 fused GEMM -> hbuf (relu)
    k_sage_gemm<<<gemm_blocks, TPB, 0, stream>>>(agg, deg, x, wT, b1, hbuf, N);

    // 6) re-zero agg
    k_zero<<<4096, TPB, 0, stream>>>(agg, aggN);

    // 7) layer-2 scatter: hbuf -> agg
    k_scatter<<<scatter_blocks, TPB, 0, stream>>>(hbuf, esrc, edst, agg, E);

    // 8) layer-2 fused GEMM -> hbuf in place (each wave reads its strip before storing it)
    k_sage_gemm<<<gemm_blocks, TPB, 0, stream>>>(agg, deg, hbuf, wT + 32768, b2, hbuf, N);

    // 9) final linear head -> d_out
    k_final<<<(N + TPB - 1) / TPB, TPB, 0, stream>>>(hbuf, Wlin, blin, (float*)d_out, N);
}
